// LocalVisiblePooling_75127567941736
// MI455X (gfx1250) — compile-verified, matrix operand order checked by reading, not run on hardware
//
#include <hip/hip_runtime.h>
#include <hip/hip_bf16.h>
#include <math.h>

#define TT 128
#define BB 2048
#define DD 1024
#define AD 1024
#define KW 8
#define LL 16
#define NB 4            // batches per block in the WMMA kernel
#define AS_STRIDE 1032  // 1024 + 8 bf16 elements of padding (bank spread)

typedef __attribute__((ext_vector_type(16))) __bf16 v16bf;
typedef __attribute__((ext_vector_type(8)))  float  v8f;

union Frag { v16bf v; uint4 q[2]; };

__device__ __forceinline__ unsigned short f2bf(float f) {
    union { float f; unsigned u; } x; x.f = f;
    unsigned r = x.u + 0x7FFFu + ((x.u >> 16) & 1u);   // round-to-nearest-even
    return (unsigned short)(r >> 16);
}

// Single-instruction V_TANH_F32 on gfx1250 if the builtin exists.
__device__ __forceinline__ float fast_tanh(float x) {
#if defined(__has_builtin)
#if __has_builtin(__builtin_amdgcn_tanhf)
    return __builtin_amdgcn_tanhf(x);
#else
    return tanhf(x);
#endif
#else
    return tanhf(x);
#endif
}

// ---------------- Kernel 0: W1 fp32 -> bf16 ----------------
__global__ void cvt_w1_kernel(const float* __restrict__ W1,
                              unsigned short* __restrict__ W1bf, int n) {
    int i = blockIdx.x * 256 + threadIdx.x;
    if (i < n) W1bf[i] = f2bf(W1[i]);
}

// ---------------- Kernel 1: per-batch s[l] via bf16 WMMA ----------------
__global__ void attn_score_kernel(const float* __restrict__ h,
                                  const int* __restrict__ offsets,
                                  const int* __restrict__ stc_lens,
                                  const int* __restrict__ sep_lst,
                                  const unsigned short* __restrict__ W1bf,
                                  const float* __restrict__ W2,
                                  float* __restrict__ s_buf) {
    extern __shared__ char smem[];
    unsigned short* As   = (unsigned short*)smem;                       // NB*16 rows x AS_STRIDE bf16
    int*            rowS = (int*)(smem + (size_t)NB * LL * AS_STRIDE * 2);
    float*          s_sh = (float*)(rowS + NB * LL);

    const int tid = threadIdx.x;
    const int b0  = blockIdx.x * NB;

    if (tid < NB * LL) {
        int j = tid >> 4, l = tid & 15;
        int b   = b0 + j;
        int off = offsets[b];
        int sep = sep_lst[b];          // (B,1) row-major
        int sl  = stc_lens[b];
        bool seg1 = (off <= sep);
        int start = seg1 ? max(off - KW, 0)   : max(off - KW, sep + 1);
        int end   = seg1 ? min(off + KW, sep) : min(off + KW, sl);
        int idx   = start + l;
        bool valid = idx < end;
        int idx_c  = min(max(idx, 0), TT - 1);
        rowS[tid] = valid ? idx_c : -1;
        s_sh[tid] = 0.f;
    }
    __syncthreads();

    // Gather blk rows into LDS as bf16 (zero for invalid rows).
    for (int e = tid; e < NB * LL * DD; e += blockDim.x) {
        int j = e >> 14;             // / (16*1024)
        int l = (e >> 10) & 15;
        int d = e & (DD - 1);
        int src = rowS[(j << 4) + l];
        float v = 0.f;
        if (src >= 0) v = h[((size_t)src * BB + (b0 + j)) * DD + d];
        As[((j << 4) + l) * AS_STRIDE + d] = f2bf(v);
    }
    __syncthreads();

    const int lane   = tid & 31;
    const int wave   = tid >> 5;
    const int laneHi = (lane >> 4) & 1;   // 0: lanes 0-15, 1: lanes 16-31
    const int lcol   = lane & 15;

    float p[NB][8];
    #pragma unroll
    for (int j = 0; j < NB; ++j)
        #pragma unroll
        for (int i = 0; i < 8; ++i) p[j][i] = 0.f;

    const v8f zero8 = {0.f,0.f,0.f,0.f,0.f,0.f,0.f,0.f};

    for (int t = 0; t < 8; ++t) {                 // 8 N-tiles per wave, 64 total
        const int n0 = (wave * 8 + t) * 16;
        v8f acc[NB];
        #pragma unroll
        for (int j = 0; j < NB; ++j) acc[j] = zero8;

        for (int k0 = 0; k0 < DD; k0 += 32) {
            // B fragment: W1^T tile, lane lcol supplies W1 row (n0+lcol)
            Frag bf;
            const unsigned short* wp =
                W1bf + (size_t)(n0 + lcol) * DD + k0 + laneHi * 8;
            bf.q[0] = *(const uint4*)(wp);
            bf.q[1] = *(const uint4*)(wp + 16);
            #pragma unroll
            for (int j = 0; j < NB; ++j) {
                Frag af;  // A fragment from LDS: row lcol of batch j's blk
                const unsigned short* ap =
                    As + ((j << 4) + lcol) * AS_STRIDE + k0 + laneHi * 8;
                af.q[0] = *(const uint4*)(ap);
                af.q[1] = *(const uint4*)(ap + 16);
                acc[j] = __builtin_amdgcn_wmma_f32_16x16x32_bf16(
                    false, af.v, false, bf.v, (short)0, acc[j], false, false);
            }
        }
        const float w2v = W2[n0 + lcol];
        #pragma unroll
        for (int j = 0; j < NB; ++j)
            #pragma unroll
            for (int i = 0; i < 8; ++i)
                p[j][i] += fast_tanh(acc[j][i]) * w2v;
    }

    // Reduce over N (lanes) within each 16-lane half.
    #pragma unroll
    for (int m = 1; m <= 8; m <<= 1)
        #pragma unroll
        for (int j = 0; j < NB; ++j)
            #pragma unroll
            for (int i = 0; i < 8; ++i)
                p[j][i] += __shfl_xor(p[j][i], m, 32);

    if (lcol == 0) {
        #pragma unroll
        for (int j = 0; j < NB; ++j)
            #pragma unroll
            for (int i = 0; i < 8; ++i)
                atomicAdd(&s_sh[(j << 4) + i + laneHi * 8], p[j][i]);
    }
    __syncthreads();

    if (tid < NB * LL) {
        int j = tid >> 4, l = tid & 15;
        s_buf[l * BB + b0 + j] = s_sh[tid];
    }
}

// ---------------- Kernel 2: z[l] = logsumexp_b s[l,b] ----------------
__global__ void lse_kernel(const float* __restrict__ s_buf, float* __restrict__ z) {
    __shared__ float red[256];
    const int l = blockIdx.x, tid = threadIdx.x;
    float m = -3.4e38f;
    for (int b = tid; b < BB; b += 256) m = fmaxf(m, s_buf[l * BB + b]);
    red[tid] = m; __syncthreads();
    for (int s = 128; s > 0; s >>= 1) {
        if (tid < s) red[tid] = fmaxf(red[tid], red[tid + s]);
        __syncthreads();
    }
    const float mg = red[0]; __syncthreads();
    float sum = 0.f;
    for (int b = tid; b < BB; b += 256) sum += expf(s_buf[l * BB + b] - mg);
    red[tid] = sum; __syncthreads();
    for (int s = 128; s > 0; s >>= 1) {
        if (tid < s) red[tid] += red[tid + s];
        __syncthreads();
    }
    if (tid == 0) z[l] = mg + logf(red[0]);
}

// ---------------- Kernel 3: softmax over l + weighted pooling ----------------
__global__ void pool_kernel(const float* __restrict__ h,
                            const int* __restrict__ offsets,
                            const int* __restrict__ stc_lens,
                            const int* __restrict__ sep_lst,
                            const float* __restrict__ s_buf,
                            const float* __restrict__ z,
                            float* __restrict__ out) {
    __shared__ float w_sh[LL];
    __shared__ int   src_sh[LL];
    const int b = blockIdx.x, tid = threadIdx.x;
    if (tid < LL) {
        int off = offsets[b], sep = sep_lst[b], sl = stc_lens[b];
        bool seg1 = (off <= sep);
        int start = seg1 ? max(off - KW, 0)   : max(off - KW, sep + 1);
        int end   = seg1 ? min(off + KW, sep) : min(off + KW, sl);
        int idx   = start + tid;
        bool valid = idx < end;
        src_sh[tid] = valid ? min(max(idx, 0), TT - 1) : -1;
        w_sh[tid]   = valid ? (s_buf[tid * BB + b] - z[tid]) : -3.4e38f;
    }
    __syncthreads();
    if (tid == 0) {
        float m = -3.4e38f;
        for (int l = 0; l < LL; ++l) m = fmaxf(m, w_sh[l]);
        float e[LL], sum = 0.f;
        for (int l = 0; l < LL; ++l) {
            e[l] = (w_sh[l] > -3.0e38f) ? expf(w_sh[l] - m) : 0.f;
            sum += e[l];
        }
        float inv = 1.f / sum;
        for (int l = 0; l < LL; ++l) w_sh[l] = e[l] * inv;
    }
    __syncthreads();
    for (int d = tid; d < DD; d += blockDim.x) {
        float acc = 0.f;
        #pragma unroll
        for (int l = 0; l < LL; ++l) {
            int src = src_sh[l];
            if (src >= 0) acc += w_sh[l] * h[((size_t)src * BB + b) * DD + d];
        }
        out[(size_t)b * DD + d] = acc;
    }
}

extern "C" void kernel_launch(void* const* d_in, const int* in_sizes, int n_in,
                              void* d_out, int out_size, void* d_ws, size_t ws_size,
                              hipStream_t stream) {
    const float* h        = (const float*)d_in[0];
    const int*   offsets  = (const int*)d_in[1];
    const int*   stc_lens = (const int*)d_in[2];
    const int*   sep_lst  = (const int*)d_in[3];
    /* d_in[4] = no_local (unused by reference) */
    const float* W1       = (const float*)d_in[5];
    const float* W2       = (const float*)d_in[6];
    float*       out      = (float*)d_out;

    char* ws = (char*)d_ws;
    unsigned short* W1bf = (unsigned short*)ws;                    // 2 MB
    float* s_buf = (float*)(ws + (size_t)AD * DD * sizeof(unsigned short));
    float* zbuf  = s_buf + (size_t)LL * BB;

    cvt_w1_kernel<<<(AD * DD) / 256, 256, 0, stream>>>(W1, W1bf, AD * DD);

    size_t lds = (size_t)NB * LL * AS_STRIDE * 2 + NB * LL * 4 + NB * LL * 4;
    (void)hipFuncSetAttribute((const void*)attn_score_kernel,
                              hipFuncAttributeMaxDynamicSharedMemorySize, (int)lds);
    attn_score_kernel<<<BB / NB, 256, lds, stream>>>(h, offsets, stc_lens, sep_lst,
                                                     W1bf, W2, s_buf);

    lse_kernel<<<LL, 256, 0, stream>>>(s_buf, zbuf);

    pool_kernel<<<BB, 256, 0, stream>>>(h, offsets, stc_lens, sep_lst,
                                        s_buf, zbuf, out);
}
